// GPT_80229989089832
// MI455X (gfx1250) — matrix operand order repaired and checked by reading, not compiled, not run on hardware
//
#include <hip/hip_runtime.h>

typedef __attribute__((ext_vector_type(16))) __bf16 v16bf;
typedef __attribute__((ext_vector_type(8)))  __bf16 v8bf;
typedef __attribute__((ext_vector_type(8)))  float  v8f;

#define N_EMBD   384
#define N_HEADS  6
#define HEAD_DIM 64
#define N_LAYER  6
#define SEQ      256
#define NBATCH   64
#define DFF      1536
#define NVOCAB   65
#define MROWS    (NBATCH * SEQ)   // 16384

static __device__ __forceinline__ __bf16 f2bf(float f) { return (__bf16)f; }

// ---------------------------------------------------------------------------
// Embedding: x[b,t,:] = tok_emb[idx[b,t],:] + pos_emb[t,:]
// ---------------------------------------------------------------------------
__global__ __launch_bounds__(256) void embed_kernel(
    const int* __restrict__ idx, const float* __restrict__ tok,
    const float* __restrict__ pos, float* __restrict__ x)
{
    int i = blockIdx.x * 256 + threadIdx.x;
    int c  = i % N_EMBD;
    int bt = i / N_EMBD;
    int t  = bt % SEQ;
    x[i] = tok[idx[bt] * N_EMBD + c] + pos[t * N_EMBD + c];
}

// ---------------------------------------------------------------------------
// LayerNorm: one wave (32 lanes) per 384-wide row, 8 rows per block
// ---------------------------------------------------------------------------
__global__ __launch_bounds__(256) void ln_kernel(
    const float* __restrict__ x, const float* __restrict__ g,
    const float* __restrict__ b, float* __restrict__ h)
{
    int wave = threadIdx.x >> 5;
    int lane = threadIdx.x & 31;
    int row  = blockIdx.x * 8 + wave;
    const float* xr = x + row * N_EMBD;
    float v[12];
    float s = 0.f;
#pragma unroll
    for (int i = 0; i < 12; ++i) { v[i] = xr[lane + 32 * i]; s += v[i]; }
#pragma unroll
    for (int off = 16; off; off >>= 1) s += __shfl_xor(s, off, 32);
    float mu = s * (1.0f / N_EMBD);
    float var = 0.f;
#pragma unroll
    for (int i = 0; i < 12; ++i) { float d = v[i] - mu; var += d * d; }
#pragma unroll
    for (int off = 16; off; off >>= 1) var += __shfl_xor(var, off, 32);
    var *= (1.0f / N_EMBD);
    float rstd = rsqrtf(var + 1e-5f);
    float* hr = h + row * N_EMBD;
#pragma unroll
    for (int i = 0; i < 12; ++i) {
        int c = lane + 32 * i;
        hr[c] = (v[i] - mu) * rstd * g[c] + b[c];
    }
}

// ---------------------------------------------------------------------------
// WMMA GEMM (N, K compile-time): C[M,N] = A[M,K] @ B[K,N] (+bias)(+res)(ReLU)
// Block = 256 thr = 8 waves covering 64 rows x 128 cols of C.
// A k-panel (64x32, shared by all waves) staged in LDS as bf16 in WMMA
// A-fragment order (double buffered); B columns per-wave disjoint, loaded as
// one clause of 16 b32 with constant immediate offsets (e*N*4).  f32 accum.
// ---------------------------------------------------------------------------
template <int N, int K>
__global__ __launch_bounds__(256) void gemm_kernel(
    const float* __restrict__ A, const float* __restrict__ Bm,
    const float* __restrict__ bias, const float* res,
    float* C, int relu)
{
    __shared__ __align__(64) __bf16 aTile[2][64 * 32];

    int tid  = threadIdx.x;
    int wave = tid >> 5;
    int lane = tid & 31;
    int hi = lane >> 4, ln = lane & 15;
    int rowBase = blockIdx.y * 64;
    int col  = blockIdx.x * 128 + wave * 16 + ln;
    int colL = (N % 128 == 0) ? col : min(col, N - 1);  // folds away when exact
    bool colOK = (N % 128 == 0) ? true : (col < N);
    int kbB = hi * 16;

    // Cooperative A loader: thread t handles row (t>>2), k-octet (t&3)*8,
    // stored in fragment order so wave-side reads are contiguous 32B.
    int am  = tid >> 2;
    int akg = (tid & 3) * 8;
    int aOff = am * 32 + akg;
    const float* aSrc = A + (rowBase + am) * K + akg;

    v8f acc[4] = {};
    constexpr int nk = K / 32;
    for (int ks = 0; ks < nk; ++ks) {
        int k0  = ks * 32;
        int buf = ks & 1;

        // ---- issue B loads first (independent of LDS barrier)
        const float* bRow = Bm + (k0 + kbB) * N + colL;  // + e*N : const offsets
        float bf[16];
#pragma unroll
        for (int e = 0; e < 16; ++e) bf[e] = bRow[e * N];

        // ---- stage A panel (f32 -> bf16, fragment order)
        {
            const float* p = aSrc + k0;
            v8bf w;
#pragma unroll
            for (int e = 0; e < 8; ++e) w[e] = f2bf(p[e]);
            *(v8bf*)&aTile[buf][aOff] = w;
        }
        __syncthreads();   // fences LDS: write visibility + WAR on buf

        v16bf bb;
#pragma unroll
        for (int e = 0; e < 16; ++e) bb[e] = f2bf(bf[e]);
        if (ks + 1 < nk)
            __builtin_prefetch(&bRow[32 * N], 0, 1);

        // ---- 4 m-tile fragments from LDS, then 4 WMMAs
        v16bf afr[4];
#pragma unroll
        for (int mt = 0; mt < 4; ++mt)
            afr[mt] = *(const v16bf*)&aTile[buf][(mt * 16 + ln) * 32 + hi * 16];
#pragma unroll
        for (int mt = 0; mt < 4; ++mt)
            acc[mt] = __builtin_amdgcn_wmma_f32_16x16x32_bf16(
                false, afr[mt], false, bb, (short)0, acc[mt], false, false);
    }

    if (!colOK) return;
    float bval = bias ? bias[col] : 0.f;
#pragma unroll
    for (int mt = 0; mt < 4; ++mt) {
#pragma unroll
        for (int r = 0; r < 8; ++r) {
            int orow = rowBase + mt * 16 + r + 8 * hi;
            float vv = acc[mt][r] + bval;
            if (res)  vv += res[orow * N + col];
            if (relu) vv = fmaxf(vv, 0.f);
            C[orow * N + col] = vv;
        }
    }
}

// ---------------------------------------------------------------------------
// Causal flash attention per (batch, head). Block = 8 waves; grid.x = B*H*2;
// each block covers 128 query rows, wave w owns 16 rows.  Online softmax,
// O accumulated in registers; P re-laid out D-frag -> A-frag through LDS.
// q/k/v/o layout: [B*T, H*D] row-major (GEMM output layout).
// ---------------------------------------------------------------------------
__global__ __launch_bounds__(256) void attn_kernel(
    const float* __restrict__ q, const float* __restrict__ k,
    const float* __restrict__ v, float* __restrict__ o)
{
    __shared__ __align__(64) __bf16 pbuf[8][16 * 32];
    int wave = threadIdx.x >> 5;
    int lane = threadIdx.x & 31;
    int hi = lane >> 4, ln = lane & 15;
    int bh = blockIdx.x >> 1;
    int b  = bh / N_HEADS, hh = bh % N_HEADS;
    int qRowBase = (blockIdx.x & 1) * 128 + wave * 16;
    const int base = b * SEQ * N_EMBD + hh * HEAD_DIM;
    const float scale = 0.05103103630798288f;  // 384^-0.5 (reference scales by C)

    // Q fragments for the two 32-wide d chunks (reused across all s chunks)
    v16bf qa[2];
    {
        const float* Qp = q + base + (qRowBase + ln) * N_EMBD;
#pragma unroll
        for (int d0 = 0; d0 < 2; ++d0)
#pragma unroll
            for (int e = 0; e < 8; ++e) {
                qa[d0][e]     = f2bf(Qp[d0 * 32 + hi * 16 + e]);
                qa[d0][8 + e] = f2bf(Qp[d0 * 32 + hi * 16 + 8 + e]);
            }
    }

    float mOld[8], sOld[8];
    v8f oacc[4] = {};
#pragma unroll
    for (int r = 0; r < 8; ++r) { mOld[r] = -1e30f; sOld[r] = 0.f; }

    int lastS = qRowBase + 15;
    for (int sBase = 0; sBase <= lastS; sBase += 32) {
        // ---- S = Q K^T for two 16-wide s tiles (K-dim = 64 in two chunks)
        v8f s0 = {}, s1 = {};
#pragma unroll
        for (int d0 = 0; d0 < 2; ++d0) {
            v16bf kb0, kb1;
            const float* Kp0 = k + base + (sBase + ln) * N_EMBD      + d0 * 32 + hi * 16;
            const float* Kp1 = k + base + (sBase + 16 + ln) * N_EMBD + d0 * 32 + hi * 16;
#pragma unroll
            for (int e = 0; e < 16; ++e) { kb0[e] = f2bf(Kp0[e]); kb1[e] = f2bf(Kp1[e]); }
            s0 = __builtin_amdgcn_wmma_f32_16x16x32_bf16(false, qa[d0], false, kb0, (short)0, s0, false, false);
            s1 = __builtin_amdgcn_wmma_f32_16x16x32_bf16(false, qa[d0], false, kb1, (short)0, s1, false, false);
        }

        // ---- scale, causal mask, online softmax (per-row stats in regs)
        int c0 = sBase + ln, c1 = sBase + 16 + ln;
        float rowAlpha[8];
#pragma unroll
        for (int r = 0; r < 8; ++r) {
            int qrow = qRowBase + r + 8 * hi;
            float a0 = (c0 <= qrow) ? s0[r] * scale : -1e30f;
            float a1 = (c1 <= qrow) ? s1[r] * scale : -1e30f;
            float m = fmaxf(a0, a1);
#pragma unroll
            for (int off = 1; off < 16; off <<= 1) m = fmaxf(m, __shfl_xor(m, off, 32));
            float mNew = fmaxf(mOld[r], m);
            float p0 = __expf(a0 - mNew), p1 = __expf(a1 - mNew);
            float rs = p0 + p1;
#pragma unroll
            for (int off = 1; off < 16; off <<= 1) rs += __shfl_xor(rs, off, 32);
            float alpha = __expf(mOld[r] - mNew);
            sOld[r] = sOld[r] * alpha + rs;
            mOld[r] = mNew;
            rowAlpha[r] = alpha;
            s0[r] = p0; s1[r] = p1;
        }
#pragma unroll
        for (int t = 0; t < 4; ++t)
#pragma unroll
            for (int r = 0; r < 8; ++r) oacc[t][r] *= rowAlpha[r];

        // ---- D-frag P -> LDS -> A-frag P (bf16), wave-internal only
        __bf16* pw = pbuf[wave];
#pragma unroll
        for (int r = 0; r < 8; ++r) {
            int m = r + 8 * hi;
            pw[m * 32 + ln]      = f2bf(s0[r]);
            pw[m * 32 + 16 + ln] = f2bf(s1[r]);
        }
        asm volatile("s_wait_dscnt 0" ::: "memory");
        v16bf pa;
#pragma unroll
        for (int e = 0; e < 8; ++e) {
            pa[e]     = pw[ln * 32 + hi * 16 + e];
            pa[8 + e] = pw[ln * 32 + hi * 16 + 8 + e];
        }
        asm volatile("s_wait_dscnt 0" ::: "memory");

        // ---- O += P @ V  (4 d-tiles of 16)
#pragma unroll
        for (int t = 0; t < 4; ++t) {
            v16bf vb;
            const float* Vp = v + base + (sBase + hi * 16) * N_EMBD + t * 16 + ln;
#pragma unroll
            for (int e = 0; e < 16; ++e) vb[e] = f2bf(Vp[e * N_EMBD]);
            oacc[t] = __builtin_amdgcn_wmma_f32_16x16x32_bf16(
                false, pa, false, vb, (short)0, oacc[t], false, false);
        }
    }

    // ---- normalize + store
#pragma unroll
    for (int r = 0; r < 8; ++r) {
        float inv = 1.0f / sOld[r];
        int orow = qRowBase + r + 8 * hi;
        float* Op = o + base + orow * N_EMBD;
#pragma unroll
        for (int t = 0; t < 4; ++t) Op[t * 16 + ln] = oacc[t][r] * inv;
    }
}

// ---------------------------------------------------------------------------
extern "C" void kernel_launch(void* const* d_in, const int* in_sizes, int n_in,
                              void* d_out, int out_size, void* d_ws, size_t ws_size,
                              hipStream_t stream)
{
    const int*   idx   = (const int*)  d_in[0];
    const float* tok   = (const float*)d_in[1];
    const float* pos   = (const float*)d_in[2];
    const float* ln1_g = (const float*)d_in[3];
    const float* ln1_b = (const float*)d_in[4];
    const float* ln2_g = (const float*)d_in[5];
    const float* ln2_b = (const float*)d_in[6];
    const float* wq    = (const float*)d_in[7];
    const float* wk    = (const float*)d_in[8];
    const float* wv    = (const float*)d_in[9];
    const float* wo    = (const float*)d_in[10];
    const float* bo    = (const float*)d_in[11];
    const float* w1    = (const float*)d_in[12];
    const float* b1    = (const float*)d_in[13];
    const float* w2    = (const float*)d_in[14];
    const float* b2    = (const float*)d_in[15];
    const float* whead = (const float*)d_in[16];
    float* out = (float*)d_out;

    float* ws = (float*)d_ws;
    const size_t szXC = (size_t)MROWS * N_EMBD;
    float* x  = ws;
    float* h  = x + szXC;          // LN out / attention out (reused)
    float* q  = h + szXC;
    float* k  = q + szXC;
    float* v  = k + szXC;
    float* ff = v + szXC;          // MROWS * DFF

    dim3 blk(256);
    dim3 gC (N_EMBD / 128, MROWS / 64);             // N=384 GEMMs
    dim3 gF (DFF    / 128, MROWS / 64);             // N=1536 GEMM
    dim3 gV ((NVOCAB + 127) / 128, MROWS / 64);     // N=65 head GEMM

    embed_kernel<<<(MROWS * N_EMBD) / 256, blk, 0, stream>>>(idx, tok, pos, x);

    for (int l = 0; l < N_LAYER; ++l) {
        const size_t wOff  = (size_t)l * N_EMBD * N_EMBD;
        const size_t w1Off = (size_t)l * N_EMBD * DFF;
        const size_t w2Off = (size_t)l * DFF * N_EMBD;

        ln_kernel<<<MROWS / 8, blk, 0, stream>>>(x, ln1_g + l * N_EMBD, ln1_b + l * N_EMBD, h);
        gemm_kernel<N_EMBD, N_EMBD><<<gC, blk, 0, stream>>>(h, wq + wOff, nullptr, nullptr, q, 0);
        gemm_kernel<N_EMBD, N_EMBD><<<gC, blk, 0, stream>>>(h, wk + wOff, nullptr, nullptr, k, 0);
        gemm_kernel<N_EMBD, N_EMBD><<<gC, blk, 0, stream>>>(h, wv + wOff, nullptr, nullptr, v, 0);

        attn_kernel<<<NBATCH * N_HEADS * 2, blk, 0, stream>>>(q, k, v, h);

        // x = x + attn_out @ wo + bo
        gemm_kernel<N_EMBD, N_EMBD><<<gC, blk, 0, stream>>>(h, wo + wOff, bo + l * N_EMBD, x, x, 0);

        ln_kernel<<<MROWS / 8, blk, 0, stream>>>(x, ln2_g + l * N_EMBD, ln2_b + l * N_EMBD, h);
        // ff = relu(h @ w1 + b1)
        gemm_kernel<DFF, N_EMBD><<<gF, blk, 0, stream>>>(h, w1 + w1Off, b1 + l * DFF, nullptr, ff, 1);
        // x = x + ff @ w2 + b2
        gemm_kernel<N_EMBD, DFF><<<gC, blk, 0, stream>>>(ff, w2 + w2Off, b2 + l * N_EMBD, x, x, 0);
    }

    // logits = x @ w_head   [16384, 65]
    gemm_kernel<NVOCAB, N_EMBD><<<gV, blk, 0, stream>>>(x, whead, nullptr, nullptr, out, 0);
}
